// LSTMPoseClassifier_56332791054728
// MI455X (gfx1250) — matrix operand order repaired and checked by reading, not compiled, not run on hardware
//
#include <hip/hip_runtime.h>
#include <hip/hip_bf16.h>

// ---------------------------------------------------------------------------
// LSTMPoseClassifier for MI455X (gfx1250), wave32 + WMMA bf16.
// B=128 T=1024 I=66 H=256 C=60, gates G4=1024.
// ---------------------------------------------------------------------------

typedef __attribute__((ext_vector_type(16))) __bf16 v16bf;
typedef __attribute__((ext_vector_type(8)))  __bf16 v8bf;
typedef __attribute__((ext_vector_type(8)))  float  v8f;

#define BB     128
#define TT     1024
#define II     66
#define HH     256
#define CCOUT  60
#define G4     1024
#define MROWS  (BB*TT)      // 131072
#define KPAD0  96
#define NTILES 64           // G4/16

// ---- gfx1250 async global->LDS copy (ASYNCcnt path), guarded --------------
#if defined(__has_builtin)
#if __has_builtin(__builtin_amdgcn_global_load_async_to_lds_b128)
#define ASYNC_LDS 1
#endif
#endif

#if defined(ASYNC_LDS)
typedef int v4i_ __attribute__((vector_size(16)));
typedef __attribute__((address_space(1))) v4i_* as1_v4i;
typedef __attribute__((address_space(3))) v4i_* as3_v4i;
#endif

__device__ __forceinline__ void copy16_g2l(const __bf16* g, __bf16* l) {
#if defined(ASYNC_LDS)
    __builtin_amdgcn_global_load_async_to_lds_b128((as1_v4i)g, (as3_v4i)l, 0, 0);
#else
    *(v8bf*)l = *(const v8bf*)g;
#endif
}

__device__ __forceinline__ void wait_async_copies() {
#if defined(ASYNC_LDS)
#if __has_builtin(__builtin_amdgcn_s_wait_asynccnt)
    __builtin_amdgcn_s_wait_asynccnt(0);
#else
    asm volatile("s_wait_asynccnt 0" ::: "memory");
#endif
#endif
}

__device__ __forceinline__ float sigmoidf_(float x) {
    return 1.0f / (1.0f + __expf(-x));
}

// Build a 16x32 bf16 A-fragment from a row-major [M][K] tile.
// Per CDNA5 ISA: lanes 0-15 -> K {0..7, 16..23}; lanes 16-31 -> K {8..15, 24..31}.
__device__ __forceinline__ v16bf load_fragA(const __bf16* rowbase, int lane) {
    const int klo = (lane < 16) ? 0 : 8;
    v8bf lo = *(const v8bf*)(rowbase + klo);
    v8bf hi = *(const v8bf*)(rowbase + klo + 16);
    return __builtin_shufflevector(lo, hi, 0,1,2,3,4,5,6,7,8,9,10,11,12,13,14,15);
}

// ---------------------------------------------------------------------------
// Cast x (f32 [131072][66]) -> bf16 [131072][96] zero-padded.
// ---------------------------------------------------------------------------
__global__ __launch_bounds__(256) void k_castx(const float* __restrict__ x,
                                               __bf16* __restrict__ xb) {
    long long idx = (long long)blockIdx.x * 256 + threadIdx.x;
    if (idx >= (long long)MROWS * KPAD0) return;
    int k   = (int)(idx % KPAD0);
    long long row = idx / KPAD0;
    float v = (k < II) ? x[row * II + k] : 0.0f;
    xb[idx] = (__bf16)v;
}

// ---------------------------------------------------------------------------
// Pack a weight matrix (f32 [2][G4][Ksrc]) into WMMA B-fragment order:
// dst[dir][kt][nt][lane][e] (bf16), e:0..15, zero-padded in K.
// ---------------------------------------------------------------------------
__global__ __launch_bounds__(512) void k_packw(const float* __restrict__ src,
                                               __bf16* __restrict__ dst,
                                               int Ksrc, int KT) {
    const int dir  = blockIdx.z;
    const int tile = blockIdx.x;          // kt*NTILES + nt
    const int nt   = tile % NTILES;
    const int kt   = tile / NTILES;
    const int tid  = threadIdx.x;         // 0..511
    const int lane = tid >> 4;
    const int e    = tid & 15;
    const int col  = nt * 16 + (lane & 15);
    const int k    = kt * 32 + ((lane < 16) ? 0 : 8) + ((e < 8) ? e : e + 8);
    float v = (k < Ksrc) ? src[(size_t)dir * G4 * Ksrc + (size_t)col * Ksrc + k] : 0.0f;
    dst[(size_t)dir * KT * NTILES * 512 + (size_t)tile * 512 + tid] = (__bf16)v;
}

// ---------------------------------------------------------------------------
// Big input GEMM:  C[dir][131072][1024] = A[131072][Ka] @ W[dir]^T + bias[dir]
// Block: 256 threads = 8 waves; block tile 64(M) x 128(N); A panel staged in
// LDS via async global->LDS copies (ASYNCcnt) when available.
// grid = (MROWS/64, G4/128, 2)
// ---------------------------------------------------------------------------
__global__ __launch_bounds__(256) void k_gemm(const __bf16* __restrict__ A,
                                              const __bf16* __restrict__ Wp,
                                              const float*  __restrict__ bias,
                                              float* __restrict__ C,
                                              int Ka) {
    extern __shared__ __bf16 As[];                 // [64][Ka+8]
    const int strideA = Ka + 8;
    const int KT      = Ka >> 5;
    const int tid  = threadIdx.x;
    const int lane = tid & 31;
    const int wv   = tid >> 5;
    const int wm   = wv & 3;                       // 4 M-sub-tiles
    const int wn   = wv >> 2;                      // 2 N-groups of 4 tiles
    const int lh   = lane >> 4;
    const int m0   = blockIdx.x * 64;

    const __bf16* Wd    = Wp   + (size_t)blockIdx.z * KT * NTILES * 512;
    const float*  biasd = bias + (size_t)blockIdx.z * G4;
    float*        Cd    = C    + (size_t)blockIdx.z * MROWS * G4;

    // Stage A panel [64][Ka] into LDS (16B chunks, async pipe if present).
    const int chunks = Ka >> 3;
    for (int idx = tid; idx < 64 * chunks; idx += 256) {
        int row = idx / chunks, cc = idx % chunks;
        copy16_g2l(&A[(size_t)(m0 + row) * Ka + cc * 8],
                   &As[row * strideA + cc * 8]);
    }
    wait_async_copies();
    __syncthreads();

    v8f acc[4];
    #pragma unroll
    for (int i = 0; i < 4; ++i) {
        float bv = biasd[(blockIdx.y * 8 + wn * 4 + i) * 16 + (lane & 15)];
        #pragma unroll
        for (int v = 0; v < 8; ++v) acc[i][v] = bv;
    }

    const __bf16* arow = As + (wm * 16 + (lane & 15)) * strideA;
    for (int kt = 0; kt < KT; ++kt) {
        v16bf a = load_fragA(arow + kt * 32, lane);
        #pragma unroll
        for (int i = 0; i < 4; ++i) {
            int nt = blockIdx.y * 8 + wn * 4 + i;
            v16bf b = *(const v16bf*)(Wd + (((size_t)kt * NTILES + nt) * 32 + lane) * 16);
            acc[i] = __builtin_amdgcn_wmma_f32_16x16x32_bf16(
                false, a, false, b, (short)0, acc[i], false, false);
        }
    }

    #pragma unroll
    for (int i = 0; i < 4; ++i) {
        int n = (blockIdx.y * 8 + wn * 4 + i) * 16 + (lane & 15);
        #pragma unroll
        for (int v = 0; v < 8; ++v)
            Cd[(size_t)(m0 + wm * 16 + v + 8 * lh) * G4 + n] = acc[i][v];
    }
}

// ---------------------------------------------------------------------------
// Recurrence: one block per (batch-tile of 16, direction). 8 waves.
// Per step: gates[16x1024] = pre[b,t,:] + h[16x256] @ w_hh^T  via WMMA,
// then LSTM cell math; h kept in LDS (bf16), c in registers.
// Batch rows are independent -> no cross-workgroup sync needed.
// pre is cold HBM traffic -> prefetch next timestep one full step ahead.
// grid = (8, 2)
// ---------------------------------------------------------------------------
__global__ __launch_bounds__(256) void k_lstm(const float*  __restrict__ pre,   // [2][131072][1024]
                                              const __bf16* __restrict__ whp,   // packed [2][8][64][32][16]
                                              float*        __restrict__ outF,  // [131072][512] (layer1)
                                              __bf16*       __restrict__ outB,  // [131072][512] (layer0)
                                              int out_is_bf16) {
    __shared__ __bf16 hbuf[16][264];               // 16 x (256+8) bf16
    const int tid  = threadIdx.x;
    const int lane = tid & 31;
    const int wv   = tid >> 5;
    const int lh   = lane >> 4;
    const int b0   = blockIdx.x * 16;
    const int dir  = blockIdx.y;

    const float*  pred = pre + (size_t)dir * MROWS * G4;
    const __bf16* wh   = whp + (size_t)dir * 8 * NTILES * 512;
    const long long pfd = dir ? -(long long)G4 : (long long)G4;  // next-t delta (floats)

    for (int i = tid; i < 16 * 264; i += 256) (&hbuf[0][0])[i] = (__bf16)0.0f;

    float cr[2][8];
    #pragma unroll
    for (int j = 0; j < 2; ++j)
        #pragma unroll
        for (int v = 0; v < 8; ++v) cr[j][v] = 0.0f;

    for (int s = 0; s < TT; ++s) {
        const int t = dir ? (TT - 1 - s) : s;

        // acc init = preactivations (bias folded in by k_gemm); prefetch t+1.
        v8f acc[4][2];
        #pragma unroll
        for (int g = 0; g < 4; ++g)
            #pragma unroll
            for (int j = 0; j < 2; ++j) {
                int col = (g * 16 + 2 * wv + j) * 16 + (lane & 15);
                #pragma unroll
                for (int v = 0; v < 8; ++v) {
                    int b = b0 + v + 8 * lh;
                    const float* p = &pred[((size_t)(b * TT + t)) * G4 + col];
                    acc[g][j][v] = *p;
                    __builtin_prefetch((const void*)(p + pfd), 0, 1);
                }
            }

        __syncthreads();   // h from previous step fully written

        #pragma unroll
        for (int kt = 0; kt < 8; ++kt) {
            v16bf a = load_fragA(&hbuf[lane & 15][kt * 32], lane);
            #pragma unroll
            for (int g = 0; g < 4; ++g)
                #pragma unroll
                for (int j = 0; j < 2; ++j) {
                    int nt = g * 16 + 2 * wv + j;
                    v16bf b = *(const v16bf*)(wh + (((size_t)kt * NTILES + nt) * 32 + lane) * 16);
                    acc[g][j] = __builtin_amdgcn_wmma_f32_16x16x32_bf16(
                        false, a, false, b, (short)0, acc[g][j], false, false);
                }
        }

        __syncthreads();   // all waves done reading h before overwrite

        #pragma unroll
        for (int j = 0; j < 2; ++j) {
            int col = (2 * wv + j) * 16 + (lane & 15);
            #pragma unroll
            for (int v = 0; v < 8; ++v) {
                float iv = acc[0][j][v];
                float fv = acc[1][j][v];
                float gv = acc[2][j][v];
                float ov = acc[3][j][v];
                float cc = sigmoidf_(fv) * cr[j][v] + sigmoidf_(iv) * tanhf(gv);
                cr[j][v] = cc;
                float hh = sigmoidf_(ov) * tanhf(cc);
                int m = v + 8 * lh;
                hbuf[m][col] = (__bf16)hh;
                size_t orow = (size_t)((b0 + m) * TT + t);
                if (out_is_bf16) outB[orow * 512 + dir * HH + col] = (__bf16)hh;
                else             outF[orow * 512 + dir * HH + col] = hh;
            }
        }
    }
}

// ---------------------------------------------------------------------------
// Stats: per-feature sum & sumsq over all 131072 rows of h1[131072][512].
// ---------------------------------------------------------------------------
__global__ __launch_bounds__(256) void k_zero_stats(float* st) {
    int i = blockIdx.x * 256 + threadIdx.x;
    if (i < 1024) st[i] = 0.0f;
}

__global__ __launch_bounds__(256) void k_stats(const float* __restrict__ h1,
                                               float* __restrict__ st) {
    const int tid = threadIdx.x;
    const long long r0 = (long long)blockIdx.x * 1024;
    float s0 = 0.f, q0 = 0.f, s1 = 0.f, q1 = 0.f;
    for (int r = 0; r < 1024; ++r) {
        float v0 = h1[(r0 + r) * 512 + tid];
        float v1 = h1[(r0 + r) * 512 + tid + 256];
        s0 += v0; q0 += v0 * v0;
        s1 += v1; q1 += v1 * v1;
    }
    atomicAdd(&st[tid],             s0);
    atomicAdd(&st[512 + tid],       q0);
    atomicAdd(&st[tid + 256],       s1);
    atomicAdd(&st[512 + tid + 256], q1);
}

// ---------------------------------------------------------------------------
// FC1 on normalized last timestep: grid=128 (batch), block=256 (out feature).
// ---------------------------------------------------------------------------
__global__ __launch_bounds__(256) void k_fc1(const float* __restrict__ h1,
                                             const float* __restrict__ st,
                                             const float* __restrict__ gamma,
                                             const float* __restrict__ beta,
                                             const float* __restrict__ fc1_w,
                                             const float* __restrict__ fc1_b,
                                             float* __restrict__ fc1o) {
    const int b = blockIdx.x, j = threadIdx.x;
    const float inv = 1.0f / (float)MROWS;
    float s = fc1_b[j];
    const size_t base = (size_t)(b * TT + (TT - 1)) * 512;
    for (int k = 0; k < 512; ++k) {
        float mean = st[k] * inv;
        float var  = st[512 + k] * inv - mean * mean;
        float rs   = rsqrtf(var + 1e-5f);
        float hn   = (h1[base + k] - mean) * rs * gamma[k] + beta[k];
        s += hn * fc1_w[(size_t)j * 512 + k];
    }
    fc1o[b * 256 + j] = fmaxf(s, 0.0f);
}

__global__ __launch_bounds__(64) void k_fc2(const float* __restrict__ fc1o,
                                            const float* __restrict__ fc2_w,
                                            const float* __restrict__ fc2_b,
                                            float* __restrict__ out) {
    const int b = blockIdx.x, j = threadIdx.x;
    if (j >= CCOUT) return;
    float s = fc2_b[j];
    for (int k = 0; k < 256; ++k)
        s += fc1o[b * 256 + k] * fc2_w[(size_t)j * 256 + k];
    out[b * CCOUT + j] = s;
}

// ---------------------------------------------------------------------------
extern "C" void kernel_launch(void* const* d_in, const int* in_sizes, int n_in,
                              void* d_out, int out_size, void* d_ws, size_t ws_size,
                              hipStream_t stream) {
    (void)in_sizes; (void)n_in; (void)out_size; (void)ws_size;
    const float* x       = (const float*)d_in[0];
    const float* w_ih_l0 = (const float*)d_in[1];
    const float* w_hh_l0 = (const float*)d_in[2];
    const float* b_l0    = (const float*)d_in[3];
    const float* w_ih_l1 = (const float*)d_in[4];
    const float* w_hh_l1 = (const float*)d_in[5];
    const float* b_l1    = (const float*)d_in[6];
    const float* gamma   = (const float*)d_in[7];
    const float* beta    = (const float*)d_in[8];
    const float* fc1_w   = (const float*)d_in[9];
    const float* fc1_b   = (const float*)d_in[10];
    const float* fc2_w   = (const float*)d_in[11];
    const float* fc2_b   = (const float*)d_in[12];

    char* ws = (char*)d_ws;
    size_t off = 0;
    auto alloc = [&](size_t bytes) -> char* {
        char* p = ws + off;
        off = (off + bytes + 255) & ~(size_t)255;
        return p;
    };

    __bf16* xb   = (__bf16*)alloc((size_t)MROWS * KPAD0 * 2);          // 25 MB
    __bf16* w0p  = (__bf16*)alloc((size_t)2 * 3  * NTILES * 512 * 2);  // w_ih_l0 packed
    __bf16* wh0p = (__bf16*)alloc((size_t)2 * 8  * NTILES * 512 * 2);  // w_hh_l0
    __bf16* w1p  = (__bf16*)alloc((size_t)2 * 16 * NTILES * 512 * 2);  // w_ih_l1
    __bf16* wh1p = (__bf16*)alloc((size_t)2 * 8  * NTILES * 512 * 2);  // w_hh_l1
    float*  pre  = (float*) alloc((size_t)2 * MROWS * G4 * 4);         // 1.07 GB (reused)
    __bf16* h0b  = (__bf16*)alloc((size_t)MROWS * 512 * 2);            // 134 MB
    float*  h1   = (float*) alloc((size_t)MROWS * 512 * 4);            // 268 MB
    float*  st   = (float*) alloc(1024 * 4);
    float*  fc1o = (float*) alloc((size_t)BB * HH * 4);

    // 1. Pack inputs / weights to bf16 (fragment order for all B operands).
    {
        long long n = (long long)MROWS * KPAD0;
        k_castx<<<(unsigned)((n + 255) / 256), 256, 0, stream>>>(x, xb);
    }
    k_packw<<<dim3(3  * NTILES, 1, 2), 512, 0, stream>>>(w_ih_l0, w0p,  II,  3);
    k_packw<<<dim3(8  * NTILES, 1, 2), 512, 0, stream>>>(w_hh_l0, wh0p, HH,  8);
    k_packw<<<dim3(16 * NTILES, 1, 2), 512, 0, stream>>>(w_ih_l1, w1p,  512, 16);
    k_packw<<<dim3(8  * NTILES, 1, 2), 512, 0, stream>>>(w_hh_l1, wh1p, HH,  8);
    k_zero_stats<<<4, 256, 0, stream>>>(st);

    // 2. Layer 0: input GEMM (both dirs) then recurrence.
    k_gemm<<<dim3(MROWS / 64, G4 / 128, 2), 256, 64 * (KPAD0 + 8) * 2, stream>>>(
        xb, w0p, b_l0, pre, KPAD0);
    k_lstm<<<dim3(8, 2), 256, 0, stream>>>(pre, wh0p, nullptr, h0b, 1);

    // 3. Layer 1: input GEMM on concatenated h0 (K=512) then recurrence.
    k_gemm<<<dim3(MROWS / 64, G4 / 128, 2), 256, 64 * (512 + 8) * 2, stream>>>(
        h0b, w1p, b_l1, pre, 512);
    k_lstm<<<dim3(8, 2), 256, 0, stream>>>(pre, wh1p, h1, nullptr, 0);

    // 4. Norm stats + head.
    k_stats<<<128, 256, 0, stream>>>(h1, st);
    k_fc1<<<BB, 256, 0, stream>>>(h1, st, gamma, beta, fc1_w, fc1_b, fc1o);
    k_fc2<<<BB, 64, 0, stream>>>(fc1o, fc2_w, fc2_b, (float*)d_out);
}
